// ActDecoder_48241072668870
// MI455X (gfx1250) — matrix operand order repaired and checked by reading, not compiled, not run on hardware
//
#include <hip/hip_runtime.h>
#include <hip/hip_bf16.h>
#include <math.h>

// ---------------------------------------------------------------------------
// Types / WMMA helpers (CDNA5 gfx1250, wave32, v_wmma_f32_16x16x32_bf16)
// ---------------------------------------------------------------------------
typedef __bf16 bf16_t;
typedef __attribute__((ext_vector_type(16))) __bf16 v16bf;
typedef __attribute__((ext_vector_type(8)))  float  v8f;

// Hardware f32 -> bf16 conversion (lowers to v_cvt_pk_bf16_f32 pairs)
__device__ __forceinline__ unsigned short f2bf_u16(float f) {
    __bf16 h = (__bf16)f;
    unsigned short u; __builtin_memcpy(&u, &h, 2); return u;
}
__device__ __forceinline__ v8f wmma_bf(v16bf a, v16bf b, v8f c) {
    return __builtin_amdgcn_wmma_f32_16x16x32_bf16(
        false, a, false, b, (short)0, c, false, false);
}

// --- A-fragment layout (16x32 bf16):
// lane L: M=L&15, hi=L>>4; elem e -> K = hi*16+e
// LDS tiles are staged in *fragment order*: for element (m, k):
//   ks=k>>5, krel=k&31, hi=(krel>>4)&1, e=krel&15
//   slot = (ks*32 + hi*16 + m)*16 + e
// so each lane's fragment is 16 contiguous bf16 (32B) -> ds_load_b128 pair.
__device__ __forceinline__ int frag_slot(int m, int k) {
    int ks = k >> 5, krel = k & 31;
    int hi = (krel >> 4) & 1;
    int e  = krel & 15;
    return (ks * 32 + hi * 16 + m) * 16 + e;
}
__device__ __forceinline__ v16bf afrag_lds(const unsigned short* As, int ks, int lane) {
    v16bf a;
    __builtin_memcpy(&a, As + (ks * 32 + lane) * 16, 32);
    return a;
}
// A-frag direct from global f32 (row-major, row m = lane&15, stride ld)
__device__ __forceinline__ v16bf afrag_global(const float* A, int ld,
                                              int kbase, int lane) {
    int m = lane & 15, hi8 = (lane >> 4) << 4;
    const float* p = A + m * ld + kbase + hi8;
    const float4* q0 = (const float4*)p;
    const float4* q1 = (const float4*)(p + 8);
    float4 x0 = q0[0], x1 = q0[1], x2 = q1[0], x3 = q1[1];
    v16bf a;
    a[0]=(__bf16)x0.x;  a[1]=(__bf16)x0.y;  a[2]=(__bf16)x0.z;  a[3]=(__bf16)x0.w;
    a[4]=(__bf16)x1.x;  a[5]=(__bf16)x1.y;  a[6]=(__bf16)x1.z;  a[7]=(__bf16)x1.w;
    a[8]=(__bf16)x2.x;  a[9]=(__bf16)x2.y;  a[10]=(__bf16)x2.z; a[11]=(__bf16)x2.w;
    a[12]=(__bf16)x3.x; a[13]=(__bf16)x3.y; a[14]=(__bf16)x3.z; a[15]=(__bf16)x3.w;
    return a;
}
// B-frag (32x16 bf16): lane N=lane&15, elem e -> K=(lane>>4)*16+e.
// p = &W[n][kbase + (lane>>4)*16] : 16 contiguous f32, 64B aligned -> 4x b128.
__device__ __forceinline__ v16bf bfrag_global(const float* p) {
    const float4* q = (const float4*)p;
    float4 x0 = q[0], x1 = q[1], x2 = q[2], x3 = q[3];
    v16bf b;
    b[0]=(__bf16)x0.x;  b[1]=(__bf16)x0.y;  b[2]=(__bf16)x0.z;  b[3]=(__bf16)x0.w;
    b[4]=(__bf16)x1.x;  b[5]=(__bf16)x1.y;  b[6]=(__bf16)x1.z;  b[7]=(__bf16)x1.w;
    b[8]=(__bf16)x2.x;  b[9]=(__bf16)x2.y;  b[10]=(__bf16)x2.z; b[11]=(__bf16)x2.w;
    b[12]=(__bf16)x3.x; b[13]=(__bf16)x3.y; b[14]=(__bf16)x3.z; b[15]=(__bf16)x3.w;
    return b;
}

__device__ __forceinline__ float sigmoidf_(float x) { return 1.0f / (1.0f + expf(-x)); }

#define V_  32000
#define E_  512
#define H_  512
#define T_  64
#define B_  16
#define VT_ 32064   // V + T

// ---------------------------------------------------------------------------
// 1) hpart[b,j] = h0 @ attn_W[:, :512].T          (16 x 512)
// grid 32 (N tiles), block 32 (1 wave)
// ---------------------------------------------------------------------------
__global__ void k_hpart(const float* __restrict__ h0,
                        const float* __restrict__ attnW,
                        float* __restrict__ hpart) {
    int nt = blockIdx.x, lane = threadIdx.x;
    int j = nt * 16 + (lane & 15);
    v8f acc0 = {0, 0, 0, 0, 0, 0, 0, 0};
    v8f acc1 = {0, 0, 0, 0, 0, 0, 0, 0};
#pragma unroll 2
    for (int ks = 0; ks < 16; ks += 2) {
        const float* bp0 = attnW + (size_t)j * (2 * H_) + ks * 32 + ((lane >> 4) << 4);
        acc0 = wmma_bf(afrag_global(h0, H_, ks * 32, lane),       bfrag_global(bp0),      acc0);
        acc1 = wmma_bf(afrag_global(h0, H_, (ks + 1) * 32, lane), bfrag_global(bp0 + 32), acc1);
    }
#pragma unroll
    for (int i = 0; i < 8; ++i) {
        int b = (lane < 16) ? i : (i + 8);
        hpart[b * H_ + j] = acc0[i] + acc1[i];
    }
}

// ---------------------------------------------------------------------------
// 2) attn_e[b,t] = sum_j tanh(enc@We.T + hpart + attn_b)[j] * attn_v[j]
// grid 64 (16-row tiles of (b,t)), block 256 (8 waves, 4 N-tiles each)
// ---------------------------------------------------------------------------
__global__ void k_attn(const float* __restrict__ enc,   // u_enc_out (T,B,H)
                       const float* __restrict__ attnW,
                       const float* __restrict__ attn_b,
                       const float* __restrict__ attn_v,
                       const float* __restrict__ hpart,
                       float* __restrict__ attn_e) {
    __shared__ unsigned short As[16 * 512];   // fragment-ordered
    __shared__ float red[16];
    int tid = threadIdx.x;
    int b  = blockIdx.x >> 2;
    int t0 = (blockIdx.x & 3) * 16;
    int r0 = blockIdx.x * 16;
    for (int idx = tid; idx < 16 * 512; idx += 256) {
        int m = idx >> 9, k = idx & 511;
        As[frag_slot(m, k)] = f2bf_u16(enc[((size_t)(t0 + m) * B_ + b) * H_ + k]);
    }
    if (tid < 16) red[tid] = 0.0f;
    __syncthreads();

    int wave = tid >> 5, lane = tid & 31;
    for (int nt = wave; nt < 32; nt += 8) {
        v8f acc0 = {0, 0, 0, 0, 0, 0, 0, 0};
        v8f acc1 = {0, 0, 0, 0, 0, 0, 0, 0};
        int j = nt * 16 + (lane & 15);
#pragma unroll 2
        for (int ks = 0; ks < 16; ks += 2) {
            const float* bp0 = attnW + (size_t)j * (2 * H_) + H_ + ks * 32 + ((lane >> 4) << 4);
            acc0 = wmma_bf(afrag_lds(As, ks, lane),     bfrag_global(bp0),      acc0);
            acc1 = wmma_bf(afrag_lds(As, ks + 1, lane), bfrag_global(bp0 + 32), acc1);
        }
        float hv = hpart[b * H_ + j] + attn_b[j];
        float av = attn_v[j];
#pragma unroll
        for (int i = 0; i < 8; ++i) {
            int m = (lane < 16) ? i : (i + 8);
            atomicAdd(&red[m], tanhf(acc0[i] + acc1[i] + hv) * av);
        }
    }
    __syncthreads();
    if (tid < 16) attn_e[r0 + tid] = red[tid];
}

// ---------------------------------------------------------------------------
// 3) alpha = softmax_T(attn_e)     grid 16, block 64
// ---------------------------------------------------------------------------
__global__ void k_softmaxT(const float* __restrict__ attn_e,
                           float* __restrict__ alpha) {
    __shared__ float s[64];
    int b = blockIdx.x, t = threadIdx.x;
    float x = attn_e[b * T_ + t];
    s[t] = x; __syncthreads();
    for (int off = 32; off > 0; off >>= 1) {
        if (t < off) s[t] = fmaxf(s[t], s[t + off]);
        __syncthreads();
    }
    float mx = s[0]; __syncthreads();
    float e = expf(x - mx);
    s[t] = e; __syncthreads();
    for (int off = 32; off > 0; off >>= 1) {
        if (t < off) s[t] += s[t + off];
        __syncthreads();
    }
    alpha[b * T_ + t] = e / s[0];
}

// ---------------------------------------------------------------------------
// 4) context[b,h] = sum_t alpha*enc ; xcat = [emb[a_tm1], context] (16x1024)
// grid 64, block 256
// ---------------------------------------------------------------------------
__global__ void k_context_xcat(const float* __restrict__ enc,
                               const float* __restrict__ alpha,
                               const float* __restrict__ emb,
                               const int*   __restrict__ a_tm1,
                               float* __restrict__ context,
                               float* __restrict__ xcat) {
    int idx = blockIdx.x * 256 + threadIdx.x;
    int b = idx >> 10, k = idx & 1023;
    if (k < H_) {
        xcat[b * 1024 + k] = emb[(size_t)a_tm1[b] * E_ + k];
    } else {
        int h = k - H_;
        float c = 0.0f;
#pragma unroll 4
        for (int t = 0; t < T_; ++t)
            c += alpha[b * T_ + t] * enc[((size_t)t * B_ + b) * H_ + h];
        context[b * H_ + h] = c;
        xcat[b * 1024 + k] = c;
    }
}

// ---------------------------------------------------------------------------
// 5) gx = xcat @ W_ih.T + b_ih ; gh = h0 @ W_hh.T + b_hh   (16 x 1536 each)
// grid (96,2), block 32
// ---------------------------------------------------------------------------
__global__ void k_gemm_gates(const float* __restrict__ xcat,
                             const float* __restrict__ h0,
                             const float* __restrict__ W_ih,
                             const float* __restrict__ b_ih,
                             const float* __restrict__ W_hh,
                             const float* __restrict__ b_hh,
                             float* __restrict__ gx,
                             float* __restrict__ gh) {
    int nt = blockIdx.x, lane = threadIdx.x;
    const float *A, *Wm, *bias; float* outp; int Kd;
    if (blockIdx.y == 0) { A = xcat; Kd = 1024; Wm = W_ih; bias = b_ih; outp = gx; }
    else                 { A = h0;   Kd = 512;  Wm = W_hh; bias = b_hh; outp = gh; }
    int n = nt * 16 + (lane & 15);
    v8f acc0 = {0, 0, 0, 0, 0, 0, 0, 0};
    v8f acc1 = {0, 0, 0, 0, 0, 0, 0, 0};
#pragma unroll 2
    for (int ks = 0; ks < Kd / 32; ks += 2) {
        const float* bp0 = Wm + (size_t)n * Kd + ks * 32 + ((lane >> 4) << 4);
        acc0 = wmma_bf(afrag_global(A, Kd, ks * 32, lane),       bfrag_global(bp0),      acc0);
        acc1 = wmma_bf(afrag_global(A, Kd, (ks + 1) * 32, lane), bfrag_global(bp0 + 32), acc1);
    }
#pragma unroll
    for (int i = 0; i < 8; ++i) {
        int b = (lane < 16) ? i : (i + 8);
        outp[b * 1536 + n] = acc0[i] + acc1[i] + bias[n];
    }
}

// ---------------------------------------------------------------------------
// 6) GRU gates -> h1 ; also writes both (1,B,H) outputs
// grid 32, block 256
// ---------------------------------------------------------------------------
__global__ void k_gru(const float* __restrict__ gx,
                      const float* __restrict__ gh,
                      const float* __restrict__ h0,
                      float* __restrict__ h1,
                      float* __restrict__ out0,
                      float* __restrict__ out1) {
    int idx = blockIdx.x * 256 + threadIdx.x;
    int b = idx >> 9, j = idx & 511;
    float r = sigmoidf_(gx[b * 1536 + j]        + gh[b * 1536 + j]);
    float z = sigmoidf_(gx[b * 1536 + 512 + j]  + gh[b * 1536 + 512 + j]);
    float n = tanhf    (gx[b * 1536 + 1024 + j] + r * gh[b * 1536 + 1024 + j]);
    float h = (1.0f - z) * n + z * h0[b * H_ + j];
    h1[idx] = h; out0[idx] = h; out1[idx] = h;
}

// ---------------------------------------------------------------------------
// 7) gen = [h1, context] @ proj_W.T + proj_b    (16 x 32000) -- streams 131MB
// grid 250, block 256 (1 N-tile per wave, 2000 tiles)
// ---------------------------------------------------------------------------
__global__ void k_proj(const float* __restrict__ h1,
                       const float* __restrict__ context,
                       const float* __restrict__ projW,
                       const float* __restrict__ proj_b,
                       float* __restrict__ gen) {
    __shared__ unsigned short Ys[16 * 1024];  // fragment-ordered
    int tid = threadIdx.x;
    for (int idx = tid; idx < 16 * 1024; idx += 256) {
        int b = idx >> 10, k = idx & 1023;
        float v = (k < H_) ? h1[b * H_ + k] : context[b * H_ + (k - H_)];
        Ys[frag_slot(b, k)] = f2bf_u16(v);
    }
    __syncthreads();
    int tile = blockIdx.x * 8 + (tid >> 5);
    int lane = tid & 31;
    int n = tile * 16 + (lane & 15);
    v8f acc0 = {0, 0, 0, 0, 0, 0, 0, 0};
    v8f acc1 = {0, 0, 0, 0, 0, 0, 0, 0};
#pragma unroll 2
    for (int ks = 0; ks < 32; ks += 2) {
        const float* bp0 = projW + (size_t)n * 1024 + ks * 32 + ((lane >> 4) << 4);
        acc0 = wmma_bf(afrag_lds(Ys, ks, lane),     bfrag_global(bp0),      acc0);
        acc1 = wmma_bf(afrag_lds(Ys, ks + 1, lane), bfrag_global(bp0 + 32), acc1);
    }
#pragma unroll
    for (int i = 0; i < 8; ++i) {
        int b = (lane < 16) ? i : (i + 8);
        gen[(size_t)b * V_ + n] = acc0[i] + acc1[i] + proj_b[n];
    }
}

// ---------------------------------------------------------------------------
// 8) ucs_raw[b,t] = sum_h tanh(enc@copy_W.T + copy_b)[h] * h1[b,h]
// grid 64, block 256
// ---------------------------------------------------------------------------
__global__ void k_copy_attn(const float* __restrict__ enc,
                            const float* __restrict__ copyW,
                            const float* __restrict__ copy_b,
                            const float* __restrict__ h1,
                            float* __restrict__ ucs_raw) {
    __shared__ unsigned short As[16 * 512];   // fragment-ordered
    __shared__ float red[16];
    int tid = threadIdx.x;
    int b  = blockIdx.x >> 2;
    int t0 = (blockIdx.x & 3) * 16;
    int r0 = blockIdx.x * 16;
    for (int idx = tid; idx < 16 * 512; idx += 256) {
        int m = idx >> 9, k = idx & 511;
        As[frag_slot(m, k)] = f2bf_u16(enc[((size_t)(t0 + m) * B_ + b) * H_ + k]);
    }
    if (tid < 16) red[tid] = 0.0f;
    __syncthreads();

    int wave = tid >> 5, lane = tid & 31;
    for (int nt = wave; nt < 32; nt += 8) {
        v8f acc0 = {0, 0, 0, 0, 0, 0, 0, 0};
        v8f acc1 = {0, 0, 0, 0, 0, 0, 0, 0};
        int j = nt * 16 + (lane & 15);
#pragma unroll 2
        for (int ks = 0; ks < 16; ks += 2) {
            const float* bp0 = copyW + (size_t)j * H_ + ks * 32 + ((lane >> 4) << 4);
            acc0 = wmma_bf(afrag_lds(As, ks, lane),     bfrag_global(bp0),      acc0);
            acc1 = wmma_bf(afrag_lds(As, ks + 1, lane), bfrag_global(bp0 + 32), acc1);
        }
        float cb = copy_b[j];
        float hm = h1[b * H_ + j];
#pragma unroll
        for (int i = 0; i < 8; ++i) {
            int m = (lane < 16) ? i : (i + 8);
            atomicAdd(&red[m], tanhf(acc0[i] + acc1[i] + cb) * hm);
        }
    }
    __syncthreads();
    if (tid < 16) ucs_raw[r0 + tid] = red[tid];
}

// ---------------------------------------------------------------------------
// 9) m[b] = max_t ucs_raw ; e_ucs = exp(ucs_raw - m)    grid 16, block 64
// ---------------------------------------------------------------------------
__global__ void k_exp_row(const float* __restrict__ ucs_raw,
                          float* __restrict__ e_ucs,
                          float* __restrict__ mrow) {
    __shared__ float s[64];
    int b = blockIdx.x, t = threadIdx.x;
    float x = ucs_raw[b * T_ + t];
    s[t] = x; __syncthreads();
    for (int off = 32; off > 0; off >>= 1) {
        if (t < off) s[t] = fmaxf(s[t], s[t + off]);
        __syncthreads();
    }
    float mx = s[0];
    e_ucs[b * T_ + t] = expf(x - mx);
    if (t == 0) mrow[b] = mx;
}

// ---------------------------------------------------------------------------
// 10) ucs_out[b,v] = log(sum_t e_ucs * sparse[b,t,v]) + m[b] -- streams 131MB
// grid 2004, block 256
// ---------------------------------------------------------------------------
__global__ void k_csum(const float* __restrict__ e_ucs,
                       const float* __restrict__ mrow,
                       const float* __restrict__ sparse,
                       float* __restrict__ ucs_out) {
    int idx = blockIdx.x * 256 + threadIdx.x;
    if (idx >= B_ * VT_) return;
    int b = idx / VT_, v = idx % VT_;
    float acc = 0.0f;
    const float* sp = sparse + (size_t)b * T_ * VT_ + v;
#pragma unroll 4
    for (int t = 0; t < T_; ++t)
        acc += e_ucs[b * T_ + t] * sp[(size_t)t * VT_];
    ucs_out[(size_t)b * VT_ + v] = logf(acc) + mrow[b];
}

// ---------------------------------------------------------------------------
// 11/12) per-b max and sum-of-exp over concat([gen(32000), ucs_out(32064)])
// grid 16, block 256 each
// ---------------------------------------------------------------------------
__global__ void k_max2(const float* __restrict__ gen,
                       const float* __restrict__ ucs_out,
                       float* __restrict__ mx) {
    __shared__ float s[256];
    int b = blockIdx.x, t = threadIdx.x;
    float m = -3.4e38f;
    for (int i = t; i < V_;  i += 256) m = fmaxf(m, gen[(size_t)b * V_ + i]);
    for (int i = t; i < VT_; i += 256) m = fmaxf(m, ucs_out[(size_t)b * VT_ + i]);
    s[t] = m; __syncthreads();
    for (int off = 128; off > 0; off >>= 1) {
        if (t < off) s[t] = fmaxf(s[t], s[t + off]);
        __syncthreads();
    }
    if (t == 0) mx[b] = s[0];
}

__global__ void k_sum2(const float* __restrict__ gen,
                       const float* __restrict__ ucs_out,
                       const float* __restrict__ mx,
                       float* __restrict__ sm) {
    __shared__ float s[256];
    int b = blockIdx.x, t = threadIdx.x;
    float m = mx[b], acc = 0.0f;
    for (int i = t; i < V_;  i += 256) acc += expf(gen[(size_t)b * V_ + i] - m);
    for (int i = t; i < VT_; i += 256) acc += expf(ucs_out[(size_t)b * VT_ + i] - m);
    s[t] = acc; __syncthreads();
    for (int off = 128; off > 0; off >>= 1) {
        if (t < off) s[t] += s[t + off];
        __syncthreads();
    }
    if (t == 0) sm[b] = s[0];
}

// ---------------------------------------------------------------------------
// 13) final probabilities (shared normalizer over 64064 logits per row)
// grid 2004, block 256
// ---------------------------------------------------------------------------
__global__ void k_proba(const float* __restrict__ gen,
                        const float* __restrict__ ucs_out,
                        const float* __restrict__ mx,
                        const float* __restrict__ sm,
                        float* __restrict__ proba) {
    int idx = blockIdx.x * 256 + threadIdx.x;
    if (idx >= B_ * VT_) return;
    int b = idx / VT_, v = idx % VT_;
    float m = mx[b], inv = 1.0f / sm[b];
    float cp = expf(ucs_out[(size_t)b * VT_ + v] - m) * inv;
    float p = (v < V_) ? (expf(gen[(size_t)b * V_ + v] - m) * inv + cp) : cp;
    proba[(size_t)b * VT_ + v] = p;
}

// ---------------------------------------------------------------------------
// Launch
// ---------------------------------------------------------------------------
extern "C" void kernel_launch(void* const* d_in, const int* in_sizes, int n_in,
                              void* d_out, int out_size, void* d_ws, size_t ws_size,
                              hipStream_t stream) {
    const float* u_enc_out = (const float*)d_in[0];   // (T,B,H)
    const int*   a_tm1     = (const int*)  d_in[1];   // (1,B)
    const float* h0        = (const float*)d_in[2];   // (1,B,H)
    const float* sparse    = (const float*)d_in[4];   // (B,T,V+T)
    const float* emb       = (const float*)d_in[5];   // (V,E)
    const float* W_ih      = (const float*)d_in[6];
    const float* b_ih      = (const float*)d_in[7];
    const float* W_hh      = (const float*)d_in[8];
    const float* b_hh      = (const float*)d_in[9];
    const float* proj_W    = (const float*)d_in[10];
    const float* proj_b    = (const float*)d_in[11];
    const float* attn_W    = (const float*)d_in[12];
    const float* attn_b    = (const float*)d_in[13];
    const float* attn_v    = (const float*)d_in[14];
    const float* copy_W    = (const float*)d_in[15];
    const float* copy_b    = (const float*)d_in[16];

    float* out = (float*)d_out;
    float* out_gru0 = out;                 // (1,B,H)
    float* out_gru1 = out + B_ * H_;       // (1,B,H)
    float* out_prob = out + 2 * B_ * H_;   // (B, V+T)

    // workspace layout (floats)
    float* W = (float*)d_ws;
    float* hpart   = W;                 // 8192
    float* attn_e  = hpart   + 8192;    // 1024
    float* alpha   = attn_e  + 1024;    // 1024
    float* context = alpha   + 1024;    // 8192
    float* xcat    = context + 8192;    // 16384
    float* gx      = xcat    + 16384;   // 24576
    float* gh      = gx      + 24576;   // 24576
    float* h1      = gh      + 24576;   // 8192
    float* ucs_raw = h1      + 8192;    // 1024
    float* e_ucs   = ucs_raw + 1024;    // 1024
    float* mrow    = e_ucs   + 1024;    // 16
    float* mx      = mrow    + 16;      // 16
    float* sm      = mx      + 16;      // 16
    float* gen     = sm      + 16;      // 512000
    float* ucs_out = gen     + 512000;  // 513024

    k_hpart       <<<32, 32, 0, stream>>>(h0, attn_W, hpart);
    k_attn        <<<64, 256, 0, stream>>>(u_enc_out, attn_W, attn_b, attn_v, hpart, attn_e);
    k_softmaxT    <<<16, 64, 0, stream>>>(attn_e, alpha);
    k_context_xcat<<<64, 256, 0, stream>>>(u_enc_out, alpha, emb, a_tm1, context, xcat);
    k_gemm_gates  <<<dim3(96, 2), 32, 0, stream>>>(xcat, h0, W_ih, b_ih, W_hh, b_hh, gx, gh);
    k_gru         <<<32, 256, 0, stream>>>(gx, gh, h0, h1, out_gru0, out_gru1);
    k_proj        <<<250, 256, 0, stream>>>(h1, context, proj_W, proj_b, gen);
    k_copy_attn   <<<64, 256, 0, stream>>>(u_enc_out, copy_W, copy_b, h1, ucs_raw);
    k_exp_row     <<<16, 64, 0, stream>>>(ucs_raw, e_ucs, mrow);
    k_csum        <<<2004, 256, 0, stream>>>(e_ucs, mrow, sparse, ucs_out);
    k_max2        <<<16, 256, 0, stream>>>(gen, ucs_out, mx);
    k_sum2        <<<16, 256, 0, stream>>>(gen, ucs_out, mx, sm);
    k_proba       <<<2004, 256, 0, stream>>>(gen, ucs_out, mx, sm, out_prob);
}
